// Qwen35Attention_73933567034081
// MI455X (gfx1250) — matrix operand order repaired and checked by reading, not compiled, not run on hardware
//
#include <hip/hip_runtime.h>
#include <hip/hip_bf16.h>

#define S 2048
#define D 2048
#define H 16
#define KVH 4
#define HD 128
#define RDIM 64
#define NQG (H * 2 * HD) // 4096
#define SCALE 0.08838834764831845f // 1/sqrt(128)

#if __has_builtin(__builtin_amdgcn_global_load_async_to_lds_b128)
#define ASYNC_LDS 1
#else
#define ASYNC_LDS 0
#endif
#if __has_builtin(__builtin_amdgcn_s_wait_asynccnt)
#define WAIT_ASYNC(n) __builtin_amdgcn_s_wait_asynccnt(n)
#else
#define WAIT_ASYNC(n)
#endif

typedef __attribute__((ext_vector_type(16))) __bf16 v16bf;
typedef __attribute__((ext_vector_type(8)))  float  v8f;
typedef __attribute__((__vector_size__(4 * sizeof(int)))) int vi4;

union BF16x16 { uint4 u4[2]; v16bf v; };

// Load a 16-element bf16 fragment from two contiguous 8-element (16B) segments.
__device__ __forceinline__ v16bf load_frag(const __bf16* __restrict__ p0,
                                           const __bf16* __restrict__ p1) {
  BF16x16 t;
  t.u4[0] = *(const uint4*)p0;
  t.u4[1] = *(const uint4*)p1;
  return t.v;
}

__device__ __forceinline__ v8f wmma_bf16(v16bf a, v16bf b, v8f c) {
  return __builtin_amdgcn_wmma_f32_16x16x32_bf16(false, a, false, b, (short)0, c, false, false);
}

// A-fragment (16x32, MxK): lane L (row M=L%16); lanes<16 hold K {0..7,16..23},
// lanes>=16 hold K {8..15,24..31}.  klo = hi?8:0.
__device__ __forceinline__ v16bf load_a(const __bf16* __restrict__ row, int k0, int klo) {
  return load_frag(row + k0 + klo, row + k0 + 16 + klo);
}
// B-fragment (32x16, KxN): lane = column N; lanes<16 hold K {0..7,16..23},
// lanes>=16 hold K {8..15,24..31} (same order as A).
__device__ __forceinline__ v16bf load_b(const __bf16* __restrict__ row, int k0, bool hi) {
  const __bf16* p = row + k0 + (hi ? 8 : 0);
  return load_frag(p, p + 16);
}

#define APAD 40  // LDS row pitch (elements): 80B -> 16B aligned, bank-conflict free

// Stage one 32x32 bf16 A tile into LDS: 128 threads x 16B segments.
// src points at (row0, k0) of a row-major (.,D) bf16 matrix.
__device__ __forceinline__ void fill_tile(const __bf16* __restrict__ src,
                                          __bf16 (* __restrict__ dst)[APAD], int tid) {
  const int row = tid >> 2;
  const int seg = tid & 3;
  const __bf16* gp = src + (size_t)row * D + seg * 8;
  __bf16* lp = &dst[row][seg * 8];
#if ASYNC_LDS
  __builtin_amdgcn_global_load_async_to_lds_b128(
      (__attribute__((address_space(1))) vi4*)gp,
      (__attribute__((address_space(3))) vi4*)lp, 0, 0);
#else
  *(uint4*)lp = *(const uint4*)gp;
#endif
}

// ------------- Conversion: f32 -> bf16, 4 elements/thread, packed store -----
__global__ void k_cvt_bf16(const float* __restrict__ in, __bf16* __restrict__ out, int n) {
  const int i = (blockIdx.x * blockDim.x + threadIdx.x) * 4;
  if (i < n) {
    const float4 f = *(const float4*)(in + i);
    union { __bf16 b[4]; uint2 u; } o;
    o.b[0] = (__bf16)f.x; o.b[1] = (__bf16)f.y;
    o.b[2] = (__bf16)f.z; o.b[3] = (__bf16)f.w;
    *(uint2*)(out + i) = o.u;
  }
}

// ------------- V transpose: (KVH,S,HD) f32 -> (KVH,HD,S) bf16, LDS tiled ----
__global__ void k_vt(const float* __restrict__ Vc, __bf16* __restrict__ Vt) {
  __shared__ float tile[32][33];
  const int kv = blockIdx.z;
  const int t0 = blockIdx.x * 32;
  const int d0 = blockIdx.y * 32;
  const float* src = Vc + (size_t)kv * S * HD;
  __bf16* dst = Vt + (size_t)kv * HD * S;
  for (int r = threadIdx.y; r < 32; r += 8)
    tile[r][threadIdx.x] = src[(size_t)(t0 + r) * HD + d0 + threadIdx.x];
  __syncthreads();
  for (int r = threadIdx.y; r < 32; r += 8)
    dst[(size_t)(d0 + r) * S + t0 + threadIdx.x] = (__bf16)tile[threadIdx.x][r];
}

// ------------- Kernel 1: QG = x @ Wq^T (M=2048,N=4096,K=2048), bf16 inputs --
// Block (4 waves) shares one 32-row A tile staged in LDS via async copy with
// double buffering; each wave owns a 64-wide N group (32x64 tile, 8 WMMA/K32).
__global__ void k_gemm_qg(const __bf16* __restrict__ xb, const __bf16* __restrict__ Wqb,
                          float* __restrict__ qg) {
  __shared__ alignas(16) __bf16 sA[2][32][APAD];
  const int tid = threadIdx.x;
  const int lane = tid & 31;
  const int wave = tid >> 5;
  const int mt = blockIdx.x >> 4;             // 0..63 (shared by the 4 waves)
  const int ng = (blockIdx.x * 4 + wave) & 63;
  const bool hi = lane >= 16;
  const int ln = lane & 15;
  const int klo = hi ? 8 : 0;
  const __bf16* asrc = xb + (size_t)(mt * 32) * D;
  v8f acc[2][4] = {};

  fill_tile(asrc, sA[0], tid);
  int buf = 0;
  for (int k0 = 0; k0 < D; k0 += 32) {
    if (k0 + 32 < D) {
      fill_tile(asrc + k0 + 32, sA[buf ^ 1], tid);
      WAIT_ASYNC(1);
    } else {
      WAIT_ASYNC(0);
    }
    __syncthreads();
    const v16bf a0 = load_frag(&sA[buf][ln][klo],      &sA[buf][ln][16 + klo]);
    const v16bf a1 = load_frag(&sA[buf][16 + ln][klo], &sA[buf][16 + ln][16 + klo]);
#pragma unroll
    for (int j = 0; j < 4; ++j) {
      const __bf16* wrow = Wqb + (size_t)(ng * 64 + j * 16 + ln) * D;
      __builtin_prefetch(wrow + k0 + 128, 0, 1);
      const v16bf b = load_b(wrow, k0, hi);
      acc[0][j] = wmma_bf16(a0, b, acc[0][j]);
      acc[1][j] = wmma_bf16(a1, b, acc[1][j]);
    }
    __syncthreads();
    buf ^= 1;
  }
#pragma unroll
  for (int mi = 0; mi < 2; ++mi)
#pragma unroll
    for (int j = 0; j < 4; ++j) {
      const int n = ng * 64 + j * 16 + ln;
#pragma unroll
      for (int r = 0; r < 8; ++r) {
        const int m = mt * 32 + mi * 16 + r + (hi ? 8 : 0);
        qg[(size_t)m * NQG + n] = acc[mi][j][r];
      }
    }
}

// ------------- Kernel 2: per-head RMSNorm + partial RoPE -> q bf16 ----------
__global__ void k_rmsrope(const float* __restrict__ qg, const float* __restrict__ cosb,
                          const float* __restrict__ sinb, const float* __restrict__ qw,
                          __bf16* __restrict__ qbf) {
  __shared__ float sh[HD];
  __shared__ float red[4];
  const int bid = blockIdx.x;     // h*S + s
  const int h = bid >> 11;
  const int s = bid & (S - 1);
  const int d = threadIdx.x;      // 0..127
  const float v = qg[(size_t)s * NQG + h * 256 + d];
  float ss = v * v;
#pragma unroll
  for (int off = 16; off >= 1; off >>= 1) ss += __shfl_xor(ss, off, 32);
  if ((threadIdx.x & 31) == 0) red[threadIdx.x >> 5] = ss;
  __syncthreads();
  const float tot = red[0] + red[1] + red[2] + red[3];
  const float qn = v * rsqrtf(tot * (1.0f / HD) + 1e-6f) * (1.0f + qw[d]);
  sh[d] = qn;
  __syncthreads();
  float out;
  if (d < RDIM) {
    const float rot = (d < 32) ? -sh[d + 32] : sh[d - 32];
    out = qn * cosb[s * RDIM + d] + rot * sinb[s * RDIM + d];
  } else {
    out = qn;
  }
  qbf[((size_t)h * S + s) * HD + d] = (__bf16)out;
}

// ------------- Kernel 3: flash attention + sigmoid gate ---------------------
// One wave per (head, 16-query block). Scores computed transposed so the
// softmax reduction per query is in-lane (8 regs) + one shfl_xor(16).
__global__ void k_attn(const __bf16* __restrict__ qbf, const __bf16* __restrict__ Kb,
                       const __bf16* __restrict__ Vt, const float* __restrict__ qg,
                       __bf16* __restrict__ obf) {
  const int lane = threadIdx.x & 31;
  const int wave = threadIdx.x >> 5;
  const int task = blockIdx.x * 4 + wave;   // 16*128 = 2048 tasks
  const int h = task >> 7;
  const int qb = task & 127;
  const int kvh = h >> 2;                   // n_rep = 4
  const bool hi = lane >= 16;
  const int ln = lane & 15;
  const int q = qb * 16 + ln;               // this lane's query column
  const int klo = hi ? 8 : 0;

  // Q^T B-fragments (128 features, 4 slabs of K=32); q already bf16
  v16bf qB[4];
  {
    const __bf16* qrow = qbf + ((size_t)h * S + q) * HD;
#pragma unroll
    for (int kk = 0; kk < 4; ++kk) qB[kk] = load_b(qrow, kk * 32, hi);
  }

  v8f o[8] = {};              // O^T: 8 tiles of (16 d-rows x 16 queries)
  float m = -3.0e38f, l = 0.0f;
  const __bf16* Kbase = Kb + (size_t)kvh * S * HD;
  const __bf16* Vtb   = Vt + (size_t)kvh * HD * S;
  const int qmax = qb * 16 + 15;

  for (int t0 = 0; t0 <= qmax; t0 += 32) {
    // S^T tiles: keys [t0..t0+15] and [t0+16..t0+31] vs 16 queries
    v8f s0 = {}, s1 = {};
    const __bf16* krow0 = Kbase + (size_t)(t0 + ln) * HD;
    const __bf16* krow1 = krow0 + (size_t)16 * HD;
    __builtin_prefetch(krow0 + 32 * HD, 0, 1);
#pragma unroll
    for (int kk = 0; kk < 4; ++kk) {
      const v16bf a0 = load_a(krow0, kk * 32, klo);
      const v16bf a1 = load_a(krow1, kk * 32, klo);
      s0 = wmma_bf16(a0, qB[kk], s0);
      s1 = wmma_bf16(a1, qB[kk], s1);
    }
    // scale + causal mask + per-query running max
    float mx = -3.0e38f;
#pragma unroll
    for (int r = 0; r < 8; ++r) {
      const int key0 = t0 + r + (hi ? 8 : 0);
      const int key1 = key0 + 16;
      s0[r] = (key0 <= q) ? s0[r] * SCALE : -1.0e9f;
      s1[r] = (key1 <= q) ? s1[r] * SCALE : -1.0e9f;
      mx = fmaxf(mx, fmaxf(s0[r], s1[r]));
    }
    mx = fmaxf(mx, __shfl_xor(mx, 16, 32));
    const float m_new = fmaxf(m, mx);
    const float fac = __expf(m - m_new);
    m = m_new;
    l *= fac;
#pragma unroll
    for (int dt = 0; dt < 8; ++dt)
#pragma unroll
      for (int r = 0; r < 8; ++r) o[dt][r] *= fac;
    // P^T = exp(S^T - m), accumulate column sum per query
    float ls = 0.0f;
#pragma unroll
    for (int r = 0; r < 8; ++r) {
      s0[r] = __expf(s0[r] - m);
      s1[r] = __expf(s1[r] - m);
      ls += s0[r] + s1[r];
    }
    ls += __shfl_xor(ls, 16, 32);
    l += ls;
    // Repack P^T (32 keys x 16 queries) into a WMMA B-fragment.
    v16bf pB;
#pragma unroll
    for (int r = 0; r < 8; ++r) {
      const float p0 = __shfl_xor(s0[r], 16, 32);
      const float p1 = __shfl_xor(s1[r], 16, 32);
      pB[r]     = (__bf16)(hi ? p1 : s0[r]);   // local t = r
      pB[r + 8] = (__bf16)(hi ? s1[r] : p0);   // local t = r + 8
    }
    // O^T += V^T (d x t) @ P^T (t x q); 8 d-tiles of 16, contiguous Vt loads
#pragma unroll
    for (int dt = 0; dt < 8; ++dt) {
      const __bf16* vrow = Vtb + (size_t)(dt * 16 + ln) * S;
      const v16bf av = load_frag(vrow + t0 + 2 * klo, vrow + t0 + 2 * klo + 8);
      o[dt] = wmma_bf16(av, pB, o[dt]);
    }
  }

  // Epilogue: normalize, sigmoid gate (gate = QG[s, h*256+128+d]), pack bf16
  const float linv = 1.0f / l;
#pragma unroll
  for (int dt = 0; dt < 8; ++dt) {
#pragma unroll
    for (int r = 0; r < 8; ++r) {
      const int d = dt * 16 + r + (hi ? 8 : 0);
      const float g = qg[(size_t)q * NQG + h * 256 + 128 + d];
      const float val = o[dt][r] * linv / (1.0f + __expf(-g));
      obf[(size_t)q * (H * HD) + h * HD + d] = (__bf16)val;
    }
  }
}

// ------------- Kernel 4: y = attn_out @ Wo^T (M=N=K=2048), bf16 inputs ------
// Same async-LDS A staging as kernel 1. Tasks = 64*32 = 2048.
__global__ void k_gemm_out(const __bf16* __restrict__ obf, const __bf16* __restrict__ Wob,
                           float* __restrict__ y) {
  __shared__ alignas(16) __bf16 sA[2][32][APAD];
  const int tid = threadIdx.x;
  const int lane = tid & 31;
  const int wave = tid >> 5;
  const int mt = blockIdx.x >> 3;             // 0..63 (shared by the 4 waves)
  const int ng = (blockIdx.x * 4 + wave) & 31;
  const bool hi = lane >= 16;
  const int ln = lane & 15;
  const int klo = hi ? 8 : 0;
  const __bf16* asrc = obf + (size_t)(mt * 32) * D;
  v8f acc[2][4] = {};

  fill_tile(asrc, sA[0], tid);
  int buf = 0;
  for (int k0 = 0; k0 < D; k0 += 32) {
    if (k0 + 32 < D) {
      fill_tile(asrc + k0 + 32, sA[buf ^ 1], tid);
      WAIT_ASYNC(1);
    } else {
      WAIT_ASYNC(0);
    }
    __syncthreads();
    const v16bf a0 = load_frag(&sA[buf][ln][klo],      &sA[buf][ln][16 + klo]);
    const v16bf a1 = load_frag(&sA[buf][16 + ln][klo], &sA[buf][16 + ln][16 + klo]);
#pragma unroll
    for (int j = 0; j < 4; ++j) {
      const __bf16* wrow = Wob + (size_t)(ng * 64 + j * 16 + ln) * D;
      __builtin_prefetch(wrow + k0 + 128, 0, 1);
      const v16bf b = load_b(wrow, k0, hi);
      acc[0][j] = wmma_bf16(a0, b, acc[0][j]);
      acc[1][j] = wmma_bf16(a1, b, acc[1][j]);
    }
    __syncthreads();
    buf ^= 1;
  }
#pragma unroll
  for (int mi = 0; mi < 2; ++mi)
#pragma unroll
    for (int j = 0; j < 4; ++j) {
      const int n = ng * 64 + j * 16 + ln;
#pragma unroll
      for (int r = 0; r < 8; ++r) {
        const int m = mt * 32 + mi * 16 + r + (hi ? 8 : 0);
        y[(size_t)m * D + n] = acc[mi][j][r];
      }
    }
}

extern "C" void kernel_launch(void* const* d_in, const int* in_sizes, int n_in,
                              void* d_out, int out_size, void* d_ws, size_t ws_size,
                              hipStream_t stream) {
  const float* x    = (const float*)d_in[0];   // (1,S,D)
  const float* Kc   = (const float*)d_in[1];   // (KVH,S,HD)
  const float* Vc   = (const float*)d_in[2];   // (KVH,S,HD)
  const float* cosb = (const float*)d_in[3];   // (S,64)
  const float* sinb = (const float*)d_in[4];   // (S,64)
  // d_in[5] causal_mask, d_in[7] Wk, d_in[8] Wv, d_in[11] k_norm_w unused
  const float* Wq   = (const float*)d_in[6];   // (4096,2048)
  const float* Wo   = (const float*)d_in[9];   // (2048,2048)
  const float* qnw  = (const float*)d_in[10];  // (128,)
  float* y = (float*)d_out;                    // (1,S,D) f32

  // Workspace layout (bytes):
  //  [0,32M)   QG f32
  //  [32M,40M) q bf16 (H,S,HD)
  //  [40M,48M) gated attn out bf16 (S,H*HD)
  //  [48M,56M) x bf16
  //  [56M,72M) Wq bf16
  //  [72M,80M) Wo bf16
  //  [80M,82M) K bf16
  //  [82M,84M) V^T bf16 (KVH,HD,S)
  char* wsb = (char*)d_ws;
  float*  qg  = (float*)wsb;
  __bf16* qbf = (__bf16*)(wsb + (size_t)32 * 1024 * 1024);
  __bf16* obf = (__bf16*)(wsb + (size_t)40 * 1024 * 1024);
  __bf16* xb  = (__bf16*)(wsb + (size_t)48 * 1024 * 1024);
  __bf16* Wqb = (__bf16*)(wsb + (size_t)56 * 1024 * 1024);
  __bf16* Wob = (__bf16*)(wsb + (size_t)72 * 1024 * 1024);
  __bf16* Kb  = (__bf16*)(wsb + (size_t)80 * 1024 * 1024);
  __bf16* Vt  = (__bf16*)(wsb + (size_t)82 * 1024 * 1024);

  const int nX  = S * D;        // 4M
  const int nWq = NQG * D;      // 8M
  const int nWo = D * D;        // 4M
  const int nK  = KVH * S * HD; // 1M
  k_cvt_bf16<<<nX  / 1024, 256, 0, stream>>>(x,  xb,  nX);
  k_cvt_bf16<<<nWq / 1024, 256, 0, stream>>>(Wq, Wqb, nWq);
  k_cvt_bf16<<<nWo / 1024, 256, 0, stream>>>(Wo, Wob, nWo);
  k_cvt_bf16<<<nK  / 1024, 256, 0, stream>>>(Kc, Kb,  nK);
  k_vt<<<dim3(S / 32, HD / 32, KVH), dim3(32, 8), 0, stream>>>(Vc, Vt);

  k_gemm_qg<<<1024, 128, 0, stream>>>(xb, Wqb, qg);
  k_rmsrope<<<H * S, 128, 0, stream>>>(qg, cosb, sinb, qnw, qbf);
  k_attn<<<512, 128, 0, stream>>>(qbf, Kb, Vt, qg, obf);
  k_gemm_out<<<512, 128, 0, stream>>>(obf, Wob, y);
}
